// GINEncoder_34205119545720
// MI455X (gfx1250) — compile-verified, matrix operand order checked by reading, not run on hardware
//
#include <hip/hip_runtime.h>
#include <hip/hip_bf16.h>

#define NN 100000
#define EE 1600000
#define HH 128
#define GG 512
#define LLAYERS 3
#define LAT 64
#define SLOPE 0.2f
#define BNEPS 1e-5f
#define TROW 5                       // row tiles per wave: 16*5*1250 == NN exactly

typedef __attribute__((ext_vector_type(16))) __bf16 v16bf;
typedef __attribute__((ext_vector_type(8)))  float  v8f;

__device__ __forceinline__ float bf2f(unsigned short b) {
    union { unsigned int u; float f; } v; v.u = ((unsigned int)b) << 16; return v.f;
}
__device__ __forceinline__ unsigned short f2bf(float f) {
    union { float f; unsigned int u; } v; v.f = f;
    unsigned int u = v.u;
    u += 0x7FFFu + ((u >> 16) & 1u);   // round-to-nearest-even
    return (unsigned short)(u >> 16);
}

// ---------------- zero fill (float4) ----------------
__global__ void k_zero4(float4* __restrict__ p, int n4) {
    int i = blockIdx.x * blockDim.x + threadIdx.x;
    if (i < n4) p[i] = make_float4(0.f, 0.f, 0.f, 0.f);
}

// ---------------- f32 -> bf16 convert (x -> h_bf) ----------------
__global__ void k_cvt(const float* __restrict__ x, unsigned short* __restrict__ o, int n) {
    int i = 4 * (blockIdx.x * blockDim.x + threadIdx.x);
    if (i < n) {
        float4 v = *(const float4*)(x + i);
        o[i + 0] = f2bf(v.x); o[i + 1] = f2bf(v.y);
        o[i + 2] = f2bf(v.z); o[i + 3] = f2bf(v.w);
    }
}

// ---------------- weight prep: transpose to N-major + bf16 ----------------
__global__ void k_prepw(const float* __restrict__ W1, const float* __restrict__ W2,
                        unsigned short* __restrict__ wt1, unsigned short* __restrict__ wt2) {
    int i = blockIdx.x * blockDim.x + threadIdx.x;
    if (i >= LLAYERS * HH * HH) return;
    int l = i / (HH * HH);
    int r = i % (HH * HH);
    int nIdx = r / HH;        // output column (N)
    int k    = r % HH;        // reduction index
    wt1[i] = f2bf(W1[l * HH * HH + k * HH + nIdx]);
    wt2[i] = f2bf(W2[l * HH * HH + k * HH + nIdx]);
}

// ---------------- edge scatter: agg[dst] += h[src] ----------------
// 32 threads per edge, 4 features per thread. Gathers are bf16 (halved
// traffic); h_bf (25.6MB) and agg (51.2MB) both live in the 192MB L2, so the
// random gathers and 205M f32 atomics per layer stay on-chip.
__global__ void k_scatter(const unsigned short* __restrict__ hbf,
                          const int* __restrict__ src, const int* __restrict__ dst,
                          float* __restrict__ agg) {
    long long tid = (long long)blockIdx.x * blockDim.x + threadIdx.x;
    int e = (int)(tid >> 5);
    if (e >= EE) return;
    int fg = ((int)tid & 31) * 4;
    int s = src[e], d = dst[e];
    uint2 raw = *(const uint2*)(hbf + (long long)s * HH + fg);
    float* ap = agg + (long long)d * HH + fg;
    atomicAdd(ap + 0, bf2f((unsigned short)(raw.x & 0xFFFFu)));
    atomicAdd(ap + 1, bf2f((unsigned short)(raw.x >> 16)));
    atomicAdd(ap + 2, bf2f((unsigned short)(raw.y & 0xFFFFu)));
    atomicAdd(ap + 3, bf2f((unsigned short)(raw.y >> 16)));
}

// ---------------- u = h + agg (in place on h_bf) ----------------
__global__ void k_addcvt(unsigned short* __restrict__ hbf, const float* __restrict__ agg, int n) {
    int i = 4 * (blockIdx.x * blockDim.x + threadIdx.x);
    if (i < n) {
        float4 a = *(const float4*)(agg + i);
        hbf[i + 0] = f2bf(bf2f(hbf[i + 0]) + a.x);
        hbf[i + 1] = f2bf(bf2f(hbf[i + 1]) + a.y);
        hbf[i + 2] = f2bf(bf2f(hbf[i + 2]) + a.z);
        hbf[i + 3] = f2bf(bf2f(hbf[i + 3]) + a.w);
    }
}

// ---------------- WMMA bf16 GEMM + fused epilogue ----------------
// Block = 8 waves; block owns an 80-row strip (TROW=5 x 16-row tiles),
// wave w owns output columns [16w, 16w+16).
// Weight (B) fragments for all K=128 are loaded ONCE per wave and reused
// across the 5 row tiles -> 20 WMMAs vs 4 B-fragment loads, 5x less weight
// traffic, and 5 independent accumulator chains to cover WMMA hazards.
// Fragment layout (16-bit A 16x32, ISA 7.12.2): row = lane&15;
//   lane<16: k 0..7 & 16..23 ; lane>=16: k 8..15 & 24..31 (two b128 loads).
__global__ __launch_bounds__(256) void k_gemm(
    const unsigned short* __restrict__ A,   // [NN,128] bf16
    const unsigned short* __restrict__ Wt,  // [128(N),128(K)] bf16
    const float* __restrict__ bias,
    const float* __restrict__ g, const float* __restrict__ beta,
    const float* __restrict__ rm, const float* __restrict__ rv,
    unsigned short* __restrict__ Out,       // [NN,128] bf16
    int useBN) {
    int wave = threadIdx.x >> 5;
    int lane = threadIdx.x & 31;
    int rowBase = blockIdx.x * (16 * TROW);
    int colBase = wave * 16;
    int lrow = lane & 15;
    int hi   = lane >> 4;

    const uint4* Ap = (const uint4*)A;     // 8 bf16 per uint4; row pitch = 16 uint4
    const uint4* Bp = (const uint4*)Wt;

    // ---- load all 4 B fragments (K = 0..127) once ----
    union Frag { uint4 q[2]; v16bf v; };
    Frag fb[4];
    int brow = colBase + lrow;
#pragma unroll
    for (int kk = 0; kk < 4; kk++) {
        int bbase = brow * 16 + kk * 4 + hi;
        fb[kk].q[0] = Bp[bbase];
        fb[kk].q[1] = Bp[bbase + 2];
    }

    // ---- 5 row tiles x 4 k-chunks ----
    v8f acc[TROW];
#pragma unroll
    for (int t = 0; t < TROW; t++)
        acc[t] = (v8f){0.f, 0.f, 0.f, 0.f, 0.f, 0.f, 0.f, 0.f};

#pragma unroll
    for (int t = 0; t < TROW; t++) {
        int arow = rowBase + t * 16 + lrow;
#pragma unroll
        for (int kk = 0; kk < 4; kk++) {
            Frag fa;
            int abase = arow * 16 + kk * 4 + hi;
            fa.q[0] = Ap[abase];
            fa.q[1] = Ap[abase + 2];
            acc[t] = __builtin_amdgcn_wmma_f32_16x16x32_bf16(
                         /*neg_a=*/false, fa.v, /*neg_b=*/false, fb[kk].v,
                         /*c_mod=*/(short)0, acc[t],
                         /*reuse_a=*/false, /*reuse_b=*/false);
        }
    }

    // ---- fused epilogue: bias + LeakyReLU (+ BN eval), bf16 writeback ----
    int col = colBase + lrow;
    float bi = bias[col];
    float sc = 1.f, sh = 0.f;
    if (useBN) {
        float s = g[col] * rsqrtf(rv[col] + BNEPS);
        sc = s;
        sh = beta[col] - rm[col] * s;
    }
#pragma unroll
    for (int t = 0; t < TROW; t++) {
#pragma unroll
        for (int r = 0; r < 8; r++) {
            int m = rowBase + t * 16 + hi * 8 + r;  // D layout: lanes16-31 -> M=8..15
            float v = acc[t][r] + bi;
            v = v > 0.f ? v : SLOPE * v;            // LeakyReLU
            if (useBN) v = sc * v + sh;             // BN eval
            Out[(long long)m * HH + col] = f2bf(v);
        }
    }
}

// ---------------- global add pool ----------------
__global__ void k_pool(const unsigned short* __restrict__ hbf,
                       const int* __restrict__ batch, float* __restrict__ pooled) {
    long long tid = (long long)blockIdx.x * blockDim.x + threadIdx.x;
    int nIdx = (int)(tid >> 5);
    if (nIdx >= NN) return;
    int fg = ((int)tid & 31) * 4;
    int gId = batch[nIdx];
    uint2 raw = *(const uint2*)(hbf + (long long)nIdx * HH + fg);
    float* pp = pooled + (long long)gId * HH + fg;
    atomicAdd(pp + 0, bf2f((unsigned short)(raw.x & 0xFFFFu)));
    atomicAdd(pp + 1, bf2f((unsigned short)(raw.x >> 16)));
    atomicAdd(pp + 2, bf2f((unsigned short)(raw.y & 0xFFFFu)));
    atomicAdd(pp + 3, bf2f((unsigned short)(raw.y >> 16)));
}

// ---------------- BN(pooled) @ fcW + fcb ----------------
__global__ __launch_bounds__(64) void k_head(
    const float* __restrict__ pooled,
    const float* __restrict__ bng, const float* __restrict__ bnb,
    const float* __restrict__ bnrm, const float* __restrict__ bnrv,
    const float* __restrict__ fcW, const float* __restrict__ fcb,
    float* __restrict__ out) {
    __shared__ float pbn[HH];
    int gId = blockIdx.x, t = threadIdx.x;
    for (int k = t; k < HH; k += 64) {
        float p = pooled[gId * HH + k];
        pbn[k] = bng[k] * (p - bnrm[k]) * rsqrtf(bnrv[k] + BNEPS) + bnb[k];
    }
    __syncthreads();
    float acc = fcb[t];
#pragma unroll 4
    for (int k = 0; k < HH; k++) acc += pbn[k] * fcW[k * LAT + t];
    out[gId * LAT + t] = acc;
}

extern "C" void kernel_launch(void* const* d_in, const int* in_sizes, int n_in,
                              void* d_out, int out_size, void* d_ws, size_t ws_size,
                              hipStream_t stream) {
    (void)in_sizes; (void)n_in; (void)out_size; (void)ws_size;
    const float* x     = (const float*)d_in[0];
    const int*   ei    = (const int*)d_in[1];          // [2,E]
    const int*   batch = (const int*)d_in[2];
    const float* W1    = (const float*)d_in[3];        // [L,128,128]
    const float* b1    = (const float*)d_in[4];
    const float* g1    = (const float*)d_in[5];
    const float* beta1 = (const float*)d_in[6];
    const float* rm1   = (const float*)d_in[7];
    const float* rv1   = (const float*)d_in[8];
    const float* W2    = (const float*)d_in[9];
    const float* b2    = (const float*)d_in[10];
    const float* bng   = (const float*)d_in[11];
    const float* bnb   = (const float*)d_in[12];
    const float* bnrm  = (const float*)d_in[13];
    const float* bnrv  = (const float*)d_in[14];
    const float* fcW   = (const float*)d_in[15];
    const float* fcb   = (const float*)d_in[16];
    float* out = (float*)d_out;

    // workspace carve-up
    char* ws = (char*)d_ws;
    size_t off = 0;
    unsigned short* h_bf = (unsigned short*)(ws + off); off += (size_t)NN * HH * 2;   // 25.6 MB
    unsigned short* t_bf = (unsigned short*)(ws + off); off += (size_t)NN * HH * 2;   // 25.6 MB
    float* agg    = (float*)(ws + off); off += (size_t)NN * HH * 4;                   // 51.2 MB
    float* pooled = (float*)(ws + off); off += (size_t)GG * HH * 4;                   // 256 KB
    unsigned short* wt1 = (unsigned short*)(ws + off); off += (size_t)LLAYERS * HH * HH * 2;
    unsigned short* wt2 = (unsigned short*)(ws + off); off += (size_t)LLAYERS * HH * HH * 2;

    const int NH = NN * HH;
    const int src_off = 0, dst_off = EE;
    const int gemmBlocks = NN / (16 * TROW);   // 1250

    // weight prep + input convert
    k_prepw<<<(LLAYERS * HH * HH + 255) / 256, 256, 0, stream>>>(W1, W2, wt1, wt2);
    k_cvt<<<(NH / 4 + 255) / 256, 256, 0, stream>>>(x, h_bf, NH);

    for (int l = 0; l < LLAYERS; ++l) {
        // agg = 0
        k_zero4<<<(NH / 4 + 255) / 256, 256, 0, stream>>>((float4*)agg, NH / 4);
        // agg[dst] += h[src]
        long long sth = (long long)EE * 32;
        k_scatter<<<(unsigned int)((sth + 255) / 256), 256, 0, stream>>>(
            h_bf, ei + src_off, ei + dst_off, agg);
        // h = h + agg  (bf16, in place)
        k_addcvt<<<(NH / 4 + 255) / 256, 256, 0, stream>>>(h_bf, agg, NH);
        // t = BN(leaky(h @ W1 + b1))
        k_gemm<<<gemmBlocks, 256, 0, stream>>>(h_bf, wt1 + l * HH * HH,
                                               b1 + l * HH, g1 + l * HH, beta1 + l * HH,
                                               rm1 + l * HH, rv1 + l * HH, t_bf, 1);
        // h = leaky(t @ W2 + b2)
        k_gemm<<<gemmBlocks, 256, 0, stream>>>(t_bf, wt2 + l * HH * HH,
                                               b2 + l * HH, nullptr, nullptr,
                                               nullptr, nullptr, h_bf, 0);
    }

    // pooled = segment_sum(h, batch)
    k_zero4<<<(GG * HH / 4 + 255) / 256, 256, 0, stream>>>((float4*)pooled, GG * HH / 4);
    long long pth = (long long)NN * 32;
    k_pool<<<(unsigned int)((pth + 255) / 256), 256, 0, stream>>>(h_bf, batch, pooled);

    // out = BN(pooled) @ fcW + fcb
    k_head<<<GG, 64, 0, stream>>>(pooled, bng, bnb, bnrm, bnrv, fcW, fcb, out);
}
